// Lstm_78984448573920
// MI455X (gfx1250) — matrix operand order repaired and checked by reading, not compiled, not run on hardware
//
#include <hip/hip_runtime.h>
#include <stdint.h>

#define S_LEN 4096
#define D_DIM 1024
#define H_DIM 2048
#define G_DIM 8192   // 4*H
#define RB    32     // persistent blocks for recurrence

typedef __attribute__((ext_vector_type(16))) __bf16 v16bf;
typedef __attribute__((ext_vector_type(8)))  float  v8f;

union V16U { int4 q[2]; v16bf v; };

__device__ __forceinline__ unsigned short f2bf(float f) {
  union { float f; unsigned u; } v; v.f = f;
  unsigned u = v.u;
  unsigned r = (u + 0x7fffu + ((u >> 16) & 1u)) >> 16;   // round-to-nearest-even
  return (unsigned short)r;
}
__device__ __forceinline__ float2 bfpair(unsigned u) {
  union { unsigned u; float f; } a, b;
  a.u = u << 16; b.u = u & 0xffff0000u;
  return make_float2(a.f, b.f);
}
__device__ __forceinline__ float sigf(float x) { return 1.0f / (1.0f + __expf(-x)); }

// Async copy 16B global -> LDS (gfx1250 GLOBAL_LOAD_ASYNC_TO_LDS_B128, ASYNCcnt).
__device__ __forceinline__ void async_g2l_b128(unsigned lds_addr, const void* gptr) {
  asm volatile("global_load_async_to_lds_b128 %0, %1, off"
               :
               : "v"(lds_addr), "v"((unsigned long long)(uintptr_t)gptr)
               : "memory");
}

// ---------------- prep kernels ----------------

__global__ void lstm_gather(const long long* __restrict__ x,
                            const float* __restrict__ emb,
                            unsigned short* __restrict__ xbf) {
  const int s = blockIdx.x;
  const long long tok = x[s];
  const float* src = emb + (size_t)tok * D_DIM;
  unsigned short* dst = xbf + (size_t)s * D_DIM;
  for (int d = threadIdx.x; d < D_DIM; d += blockDim.x) dst[d] = f2bf(src[d]);
}

__global__ void lstm_cvt_bf16(const float* __restrict__ in,
                              unsigned short* __restrict__ out, int n) {
  int i = blockIdx.x * blockDim.x + threadIdx.x;
  int stride = gridDim.x * blockDim.x;
  for (; i < n; i += stride) out[i] = f2bf(in[i]);
}

__global__ void lstm_bias(const float* __restrict__ bih,
                          const float* __restrict__ bhh,
                          float* __restrict__ bsum) {
  int i = blockIdx.x * blockDim.x + threadIdx.x;
  if (i < G_DIM) bsum[i] = bih[i] + bhh[i];
}

__global__ void lstm_init(const float* __restrict__ h0,
                          unsigned short* __restrict__ hbf,
                          int* __restrict__ cnt, int* __restrict__ gen) {
  int i = blockIdx.x * blockDim.x + threadIdx.x;
  if (i < H_DIM) hbf[i] = f2bf(h0[i]);     // buffer 0 = h at step 0
  if (i == 0) { *cnt = 0; *gen = 0; }
}

// ---------------- input-projection GEMM (bf16 WMMA, f32 accum) ----------------
// P[S, 4H] = Xbf[S, D] * W_ih^T + (b_ih + b_hh)
// Workgroup tile 128x128, 8 waves in 4(M) x 2(N), wave tile 32x64 = 2x4 WMMA tiles.
// K pipeline: double-buffered LDS tiles filled with GLOBAL_LOAD_ASYNC_TO_LDS_B128,
// drained with s_wait_asynccnt; fragments read back via ds_load_b128.

__global__ __launch_bounds__(256) void lstm_xproj_wmma(
    const unsigned short* __restrict__ Xbf,
    const unsigned short* __restrict__ Wih,   // bf16, row-major [4H, D]
    const float* __restrict__ bsum,
    float* __restrict__ P) {
  __shared__ unsigned short Ash[2][128 * 32];  // [buf][m][k], 8 KB each
  __shared__ unsigned short Bsh[2][128 * 32];  // [buf][n][k], 8 KB each

  const int t    = threadIdx.x;
  const int lane = t & 31;
  const int wave = t >> 5;
  const int wm = wave & 3;            // M offset: 32 per wave-row
  const int wn = wave >> 2;           // N offset: 64 per wave-col
  const int l    = lane & 15;
  const int half = lane >> 4;

  const int Mblk = blockIdx.y * 128;
  const int Nblk = blockIdx.x * 128;

  // Cooperative tile-fill mapping: 2 chunks/thread/matrix, 8 bf16 (16 B) each.
  int eRow[2], eKK[2];
#pragma unroll
  for (int r = 0; r < 2; ++r) {
    const int e = (t + r * 256) * 8;   // element offset within 128x32 tile
    eRow[r] = e >> 5;                  // tile row (m for A, n for B)
    eKK[r]  = e & 31;                  // k offset within tile
  }

  // issue async loads of one 128x32 K-tile pair into LDS buffer `buf`
  auto issue_tile = [&](int buf, int kb) {
#pragma unroll
    for (int r = 0; r < 2; ++r) {
      const int e = (t + r * 256) * 8;
      async_g2l_b128((unsigned)(uintptr_t)&Ash[buf][e],
                     Xbf + (size_t)(Mblk + eRow[r]) * D_DIM + kb + eKK[r]);
      async_g2l_b128((unsigned)(uintptr_t)&Bsh[buf][e],
                     Wih + (size_t)(Nblk + eRow[r]) * D_DIM + kb + eKK[r]);
    }
  };

  issue_tile(0, 0);

  v8f acc[2][4] = {};

  for (int kb = 0, it = 0; kb < D_DIM; kb += 32, ++it) {
    const int cur = it & 1;
    if (kb + 32 < D_DIM) {
      issue_tile(cur ^ 1, kb + 32);
      // 8 async-load instrs in flight per wave; allow the newest 4 to remain.
      asm volatile("s_wait_asynccnt 0x4" ::: "memory");
    } else {
      asm volatile("s_wait_asynccnt 0x0" ::: "memory");
    }
    __syncthreads();

    V16U a[2], b[4];
#pragma unroll
    for (int i = 0; i < 2; ++i) {
      // A 16x32 bf16 layout: lane<16 -> K{0..7,16..23}; lane>=16 -> K{8..15,24..31}
      const unsigned short* pa = &Ash[cur][(wm * 32 + i * 16 + l) * 32 + half * 8];
      a[i].q[0] = *(const int4*)(pa);        // elems 0..7
      a[i].q[1] = *(const int4*)(pa + 16);   // elems 8..15
    }
#pragma unroll
    for (int j = 0; j < 4; ++j) {
      // B 32x16 bf16 layout: lane<16 -> K{0..7,16..23}; lane>=16 -> K{8..15,24..31}
      const unsigned short* pb = &Bsh[cur][(wn * 64 + j * 16 + l) * 32 + half * 8];
      b[j].q[0] = *(const int4*)(pb);
      b[j].q[1] = *(const int4*)(pb + 16);
    }
#pragma unroll
    for (int i = 0; i < 2; ++i)
#pragma unroll
      for (int j = 0; j < 4; ++j)
        acc[i][j] = __builtin_amdgcn_wmma_f32_16x16x32_bf16(
            false, a[i].v, false, b[j].v, (short)0, acc[i][j], false, false);

    __syncthreads();   // all reads done before next iteration overwrites buf^1
  }

  // C/D layout: VGPR r -> row (half*8 + r), col = l
#pragma unroll
  for (int i = 0; i < 2; ++i) {
#pragma unroll
    for (int j = 0; j < 4; ++j) {
      const int n = Nblk + wn * 64 + j * 16 + l;
      const float bias = bsum[n];
      const int m0 = Mblk + wm * 32 + i * 16 + half * 8;
#pragma unroll
      for (int r = 0; r < 8; ++r)
        P[(size_t)(m0 + r) * G_DIM + n] = acc[i][j][r] + bias;
    }
  }
}

// ---------------- persistent recurrent kernel ----------------
// 32 blocks x 256 threads. Block b owns h[b*64 .. b*64+63] and the matching
// i/f/g/o gate rows. One device-wide barrier per timestep; h double-buffered.

__global__ __launch_bounds__(256) void lstm_recurrent(
    const float* __restrict__ P,
    const unsigned short* __restrict__ Whh,   // bf16 [4H, H], L2-resident (33.5 MB)
    unsigned short* __restrict__ hbf,         // [2][H] bf16 double buffer
    const float* __restrict__ c0,
    float* __restrict__ out,                  // [S, H] f32
    int* __restrict__ cnt, int* __restrict__ gen) {
  __shared__ unsigned int hsh[H_DIM / 2];     // 4 KB: current h in bf16 pairs
  __shared__ float gsh[256];

  const int t = threadIdx.x;
  const int b = blockIdx.x;
  const int hbase = b * 64;
  const int gtype = t >> 6;                   // 0=i 1=f 2=g 3=o
  const int hoff  = t & 63;
  const int row   = gtype * H_DIM + hbase + hoff;   // global gate row

  const uint4* __restrict__ w4 = (const uint4*)(Whh + (size_t)row * H_DIM);

  float c = (t < 64) ? c0[hbase + t] : 0.0f;

  for (int s = 0; s < S_LEN; ++s) {
    const unsigned int* hg =
        (const unsigned int*)(hbf + (size_t)(s & 1) * H_DIM);
    for (int i = t; i < H_DIM / 2; i += 256) hsh[i] = hg[i];
    __syncthreads();

    float acc = P[(size_t)s * G_DIM + row];
    const uint4* h4 = (const uint4*)hsh;
#pragma unroll 4
    for (int k = 0; k < H_DIM / 8; ++k) {
      uint4 w = w4[k];
      uint4 h = h4[k];
      float2 wf, hf;
      wf = bfpair(w.x); hf = bfpair(h.x);
      acc = fmaf(wf.x, hf.x, acc); acc = fmaf(wf.y, hf.y, acc);
      wf = bfpair(w.y); hf = bfpair(h.y);
      acc = fmaf(wf.x, hf.x, acc); acc = fmaf(wf.y, hf.y, acc);
      wf = bfpair(w.z); hf = bfpair(h.z);
      acc = fmaf(wf.x, hf.x, acc); acc = fmaf(wf.y, hf.y, acc);
      wf = bfpair(w.w); hf = bfpair(h.w);
      acc = fmaf(wf.x, hf.x, acc); acc = fmaf(wf.y, hf.y, acc);
    }
    gsh[t] = acc;
    __syncthreads();

    if (t < 64) {
      float gi = gsh[t];
      float gf = gsh[64 + t];
      float gg = gsh[128 + t];
      float go = gsh[192 + t];
      c = sigf(gf) * c + sigf(gi) * tanhf(gg);
      float h = sigf(go) * tanhf(c);
      out[(size_t)s * H_DIM + hbase + t] = h;
      hbf[(size_t)((s + 1) & 1) * H_DIM + hbase + t] = f2bf(h);
    }
    __threadfence();
    __syncthreads();
    if (t == 0) {                              // device-wide barrier
      int g = __hip_atomic_load(gen, __ATOMIC_RELAXED, __HIP_MEMORY_SCOPE_AGENT);
      int arrived =
          __hip_atomic_fetch_add(cnt, 1, __ATOMIC_ACQ_REL, __HIP_MEMORY_SCOPE_AGENT);
      if (arrived == RB - 1) {
        __hip_atomic_store(cnt, 0, __ATOMIC_RELAXED, __HIP_MEMORY_SCOPE_AGENT);
        __hip_atomic_fetch_add(gen, 1, __ATOMIC_ACQ_REL, __HIP_MEMORY_SCOPE_AGENT);
      } else {
        while (__hip_atomic_load(gen, __ATOMIC_ACQUIRE, __HIP_MEMORY_SCOPE_AGENT) == g) {
          __builtin_amdgcn_s_sleep(1);
        }
      }
    }
    __syncthreads();
  }
}

// ---------------- host entry ----------------

extern "C" void kernel_launch(void* const* d_in, const int* in_sizes, int n_in,
                              void* d_out, int out_size, void* d_ws, size_t ws_size,
                              hipStream_t stream) {
  (void)in_sizes; (void)n_in; (void)out_size; (void)ws_size;
  const long long* x  = (const long long*)d_in[0];
  const float* emb    = (const float*)d_in[1];
  const float* W_ih   = (const float*)d_in[2];
  const float* W_hh   = (const float*)d_in[3];
  const float* b_ih   = (const float*)d_in[4];
  const float* b_hh   = (const float*)d_in[5];
  const float* h0     = (const float*)d_in[6];
  const float* c0     = (const float*)d_in[7];
  float* out = (float*)d_out;

  char* ws = (char*)d_ws;
  size_t off = 0;
  auto alloc = [&](size_t bytes) -> char* {
    char* p = ws + off;
    off = (off + bytes + 255) & ~(size_t)255;
    return p;
  };
  unsigned short* xbf  = (unsigned short*)alloc((size_t)S_LEN * D_DIM * 2);  // 8 MB
  unsigned short* wihb = (unsigned short*)alloc((size_t)G_DIM * D_DIM * 2);  // 16 MB
  unsigned short* whhb = (unsigned short*)alloc((size_t)G_DIM * H_DIM * 2);  // 32 MB
  float* bsum          = (float*)alloc((size_t)G_DIM * 4);                   // 32 KB
  float* P             = (float*)alloc((size_t)S_LEN * G_DIM * 4);           // 128 MB
  unsigned short* hbf  = (unsigned short*)alloc((size_t)2 * H_DIM * 2);      // 8 KB
  int* sync            = (int*)alloc(256);

  lstm_gather<<<S_LEN, 256, 0, stream>>>(x, emb, xbf);
  lstm_cvt_bf16<<<1024, 256, 0, stream>>>(W_ih, wihb, G_DIM * D_DIM);
  lstm_cvt_bf16<<<2048, 256, 0, stream>>>(W_hh, whhb, G_DIM * H_DIM);
  lstm_bias<<<G_DIM / 256, 256, 0, stream>>>(b_ih, b_hh, bsum);
  lstm_init<<<(H_DIM + 255) / 256, 256, 0, stream>>>(h0, hbf, sync, sync + 1);
  lstm_xproj_wmma<<<dim3(G_DIM / 128, S_LEN / 128), 256, 0, stream>>>(
      xbf, wihb, bsum, P);
  lstm_recurrent<<<RB, 256, 0, stream>>>(P, whhb, hbf, c0, out, sync, sync + 1);
}
